// GraphAttentionNetwork_82523501625847
// MI455X (gfx1250) — matrix operand order repaired and checked by reading, not compiled, not run on hardware
//
#include <hip/hip_runtime.h>
#include <hip/hip_bf16.h>

typedef __attribute__((ext_vector_type(16))) __bf16 v16bf;
typedef __attribute__((ext_vector_type(8)))  float  v8f;

constexpr int Bc = 4, Nc = 2048, IN_DIMc = 128, HIDc = 64, HEADSc = 4;
constexpr float NEGV = -1e9f;

__device__ __forceinline__ __bf16 f2bf(float f) { return (__bf16)f; }

// ---------------------------------------------------------------------------
// Projection: H = X @ W (per head), plus s = H.a1, t = H.a2.
// Stores H transposed in bf16 ([64][N] per (head,b)) for attention B-tiles.
//   X    : [B*N][K] f32 (row-major)
//   W    : [heads][K][64] f32
//   avec : [heads][128] f32   (a1 = first 64, a2 = last 64)
//   Ht   : [heads*B][64][N] bf16
//   s,t  : [heads*B][N] f32
// Block = 128 threads (4 waves); each wave owns one 16-row M-tile.
// ---------------------------------------------------------------------------
__global__ void gat_proj_kernel(const float* __restrict__ X,
                                const float* __restrict__ W,
                                const float* __restrict__ avec,
                                __hip_bfloat16* __restrict__ Ht,
                                float* __restrict__ s_out,
                                float* __restrict__ t_out,
                                int K)
{
    const int lane    = threadIdx.x & 31;
    const int wave    = threadIdx.x >> 5;
    const int head    = blockIdx.y;
    const int rowTile = blockIdx.x * 64 + wave * 16;   // global row in [0, B*N)
    const int b       = rowTile / Nc;
    const int nb      = rowTile % Nc;
    const int col     = lane & 15;
    const int half    = lane >> 4;                     // 0 or 1
    const int koff    = half * 8;                      // 16-bit A-layout K offset
    const int j16     = half * 16;                     // contiguous K offset (used for A and B)

    const float* Wh = W + (size_t)head * K * HIDc;
    const float* xr = X + (size_t)(rowTile + col) * K; // this lane's A-row

    v8f acc[4];
    #pragma unroll
    for (int nt = 0; nt < 4; ++nt)
        acc[nt] = v8f{0.f,0.f,0.f,0.f,0.f,0.f,0.f,0.f};

    for (int kb = 0; kb < K; kb += 32) {
        // A-tile: 16x32 bf16, lane<16 holds K 0..15, lane>=16 holds K 16..31
        v16bf A;
        #pragma unroll
        for (int e = 0; e < 8; ++e) {
            A[e]     = f2bf(xr[kb + j16 + e]);
            A[e + 8] = f2bf(xr[kb + j16 + 8 + e]);
        }
        #pragma unroll
        for (int nt = 0; nt < 4; ++nt) {
            // B-tile: 32x16 bf16, lane = column, lanes 0-15: K=kb..kb+15, 16-31: +16
            v16bf Bm;
            #pragma unroll
            for (int e = 0; e < 16; ++e)
                Bm[e] = f2bf(Wh[(size_t)(kb + j16 + e) * HIDc + nt * 16 + col]);
            acc[nt] = __builtin_amdgcn_wmma_f32_16x16x32_bf16(
                false, A, false, Bm, (short)0, acc[nt], false, false);
        }
    }

    // --- store H^T bf16: C layout is lane=col, VGPR v = row v (+8 for high half)
    __hip_bfloat16* Hb = Ht + (size_t)(head * Bc + b) * HIDc * Nc;
    #pragma unroll
    for (int nt = 0; nt < 4; ++nt) {
        const int d = nt * 16 + col;
        #pragma unroll
        for (int v = 0; v < 8; ++v)
            Hb[(size_t)d * Nc + nb + half * 8 + v] = __float2bfloat16(acc[nt][v]);
    }

    // --- s = H.a1, t = H.a2 : partial per lane, butterfly-reduce across 16 cols
    const float* a1 = avec + (size_t)head * 2 * HIDc;
    const float* a2 = a1 + HIDc;
    float ps[8], pt[8];
    #pragma unroll
    for (int v = 0; v < 8; ++v) { ps[v] = 0.f; pt[v] = 0.f; }
    #pragma unroll
    for (int nt = 0; nt < 4; ++nt) {
        const float w1 = a1[nt * 16 + col];
        const float w2 = a2[nt * 16 + col];
        #pragma unroll
        for (int v = 0; v < 8; ++v) {
            ps[v] += acc[nt][v] * w1;
            pt[v] += acc[nt][v] * w2;
        }
    }
    #pragma unroll
    for (int mask = 1; mask < 16; mask <<= 1) {
        #pragma unroll
        for (int v = 0; v < 8; ++v) {
            ps[v] += __shfl_xor(ps[v], mask, 32);
            pt[v] += __shfl_xor(pt[v], mask, 32);
        }
    }
    if (col == 0) {
        float* sp = s_out + (size_t)(head * Bc + b) * Nc;
        float* tp = t_out + (size_t)(head * Bc + b) * Nc;
        #pragma unroll
        for (int v = 0; v < 8; ++v) {
            sp[nb + half * 8 + v] = ps[v];
            tp[nb + half * 8 + v] = pt[v];
        }
    }
}

// ---------------------------------------------------------------------------
// Fused masked-softmax attention: out_i = sum_j softmax_j(mask(lrelu(s_i+t_j))) * H_j
// Online (flash) softmax over 32-wide j-chunks; P@H via bf16 WMMA.
//   s,t  : [heads*B][N] f32 ; Ht : [heads*B][64][N] bf16 ; adj : [B][N][N] i32
//   out  : row stride `ostride` f32, column base = head*colMul
// ---------------------------------------------------------------------------
__global__ void gat_attn_kernel(const float* __restrict__ s_in,
                                const float* __restrict__ t_in,
                                const __hip_bfloat16* __restrict__ Ht,
                                const int* __restrict__ adj,
                                float* __restrict__ out,
                                int ostride, int colMul)
{
    const int lane    = threadIdx.x & 31;
    const int wave    = threadIdx.x >> 5;
    const int b       = blockIdx.y;
    const int head    = blockIdx.z;
    const int idx     = head * Bc + b;
    const int rowTile = blockIdx.x * 64 + wave * 16;   // n-row in [0, N)
    const int col     = lane & 15;
    const int half    = lane >> 4;
    const int koff    = half * 8;
    const int j16     = half * 16;

    const float*           tp   = t_in + (size_t)idx * Nc;
    const __hip_bfloat16*  Hb   = Ht + (size_t)idx * HIDc * Nc;
    const int*             adjr = adj + (size_t)b * Nc * Nc + (size_t)(rowTile + col) * Nc;
    const float            srow = s_in[(size_t)idx * Nc + rowTile + col];

    float m = -3.0e38f;
    float l = 0.f;
    v8f acc[4];
    #pragma unroll
    for (int nt = 0; nt < 4; ++nt)
        acc[nt] = v8f{0.f,0.f,0.f,0.f,0.f,0.f,0.f,0.f};

    for (int jb = 0; jb < Nc; jb += 32) {
        // --- scores for my A-row, contiguous K order (k = 16*half + e) ---
        float ev[16];
        #pragma unroll
        for (int e = 0; e < 8; ++e) {
            {
                const int j = jb + j16 + e;
                float ee = srow + tp[j];
                ee = fmaxf(ee, 0.2f * ee);          // leaky_relu(0.2)
                ev[e] = (adjr[j] == 0) ? NEGV : ee;
            }
            {
                const int j = jb + j16 + 8 + e;
                float ee = srow + tp[j];
                ee = fmaxf(ee, 0.2f * ee);
                ev[e + 8] = (adjr[j] == 0) ? NEGV : ee;
            }
        }
        // --- online softmax: chunk max (row split across lane pair l, l+16) ---
        float cm = ev[0];
        #pragma unroll
        for (int e = 1; e < 16; ++e) cm = fmaxf(cm, ev[e]);
        cm = fmaxf(cm, __shfl_xor(cm, 16, 32));
        const float mnew  = fmaxf(m, cm);
        const float scale = __expf(m - mnew);
        m = mnew;

        float psum = 0.f;
        v16bf A;
        #pragma unroll
        for (int e = 0; e < 16; ++e) {
            const float p = __expf(ev[e] - mnew);
            psum += p;
            A[e] = f2bf(p);
        }
        psum += __shfl_xor(psum, 16, 32);
        l = l * scale + psum;

        // rescale accumulators: C-layout row = v + 8*half, scale lives in lane==row
        float rs[8];
        #pragma unroll
        for (int v = 0; v < 8; ++v) rs[v] = __shfl(scale, half * 8 + v, 32);
        #pragma unroll
        for (int nt = 0; nt < 4; ++nt)
            #pragma unroll
            for (int v = 0; v < 8; ++v) acc[nt][v] *= rs[v];

        // --- P @ H chunk: B-tiles straight from H^T (contiguous 32B per lane) ---
        #pragma unroll
        for (int nt = 0; nt < 4; ++nt) {
            const __hip_bfloat16* hp = Hb + (size_t)(nt * 16 + col) * Nc + jb + j16;
            v16bf Bm = *(const v16bf*)hp;
            acc[nt] = __builtin_amdgcn_wmma_f32_16x16x32_bf16(
                false, A, false, Bm, (short)0, acc[nt], false, false);
        }
    }

    // --- finalize: divide by row sum, write C-layout out ---
    const float rinv = 1.0f / l;
    float li[8];
    #pragma unroll
    for (int v = 0; v < 8; ++v) li[v] = __shfl(rinv, half * 8 + v, 32);

    float* ob = out + (size_t)b * Nc * ostride + (size_t)head * colMul;
    #pragma unroll
    for (int nt = 0; nt < 4; ++nt)
        #pragma unroll
        for (int v = 0; v < 8; ++v)
            ob[(size_t)(rowTile + half * 8 + v) * ostride + nt * 16 + col] =
                acc[nt][v] * li[v];
}

// ---------------------------------------------------------------------------
extern "C" void kernel_launch(void* const* d_in, const int* in_sizes, int n_in,
                              void* d_out, int out_size, void* d_ws, size_t ws_size,
                              hipStream_t stream)
{
    const float* x   = (const float*)d_in[0];   // [B,N,128]
    const int*   adj = (const int*)  d_in[1];   // [B,N,N]
    const float* Whd = (const float*)d_in[2];   // [4,128,64]
    const float* ahd = (const float*)d_in[3];   // [4,128]
    const float* Wo  = (const float*)d_in[4];   // [256,64]
    const float* ao  = (const float*)d_in[5];   // [128]
    float* out = (float*)d_out;                 // [B,N,64]

    char* ws = (char*)d_ws;
    size_t off = 0;
    auto take = [&](size_t bytes) -> char* {
        char* p = ws + off;
        off = (off + bytes + 255) & ~(size_t)255;
        return p;
    };

    __hip_bfloat16* Ht1 = (__hip_bfloat16*)take((size_t)HEADSc * Bc * HIDc * Nc * 2);
    float* s1   = (float*)take((size_t)HEADSc * Bc * Nc * 4);
    float* t1   = (float*)take((size_t)HEADSc * Bc * Nc * 4);
    float* hcat = (float*)take((size_t)Bc * Nc * HEADSc * HIDc * 4);
    __hip_bfloat16* Ht2 = (__hip_bfloat16*)take((size_t)Bc * HIDc * Nc * 2);
    float* s2   = (float*)take((size_t)Bc * Nc * 4);
    float* t2   = (float*)take((size_t)Bc * Nc * 4);
    (void)ws_size; (void)in_sizes; (void)n_in; (void)out_size;

    const dim3 blk(128);

    // Layer 1: per-head projection + attention -> hcat [B,N,256]
    gat_proj_kernel<<<dim3((Bc * Nc) / 64, HEADSc), blk, 0, stream>>>(
        x, Whd, ahd, Ht1, s1, t1, IN_DIMc);
    gat_attn_kernel<<<dim3(Nc / 64, Bc, HEADSc), blk, 0, stream>>>(
        s1, t1, Ht1, adj, hcat, HEADSc * HIDc, HIDc);

    // Layer 2: projection from hcat (K=256) + attention -> d_out [B,N,64]
    gat_proj_kernel<<<dim3((Bc * Nc) / 64, 1), blk, 0, stream>>>(
        hcat, Wo, ao, Ht2, s2, t2, HEADSc * HIDc);
    gat_attn_kernel<<<dim3(Nc / 64, Bc, 1), blk, 0, stream>>>(
        s2, t2, Ht2, adj, out, HIDc, 0);
}
